// QuixerCore_53137335386392
// MI455X (gfx1250) — matrix operand order repaired and checked by reading, not compiled, not run on hardware
//
#include <hip/hip_runtime.h>
#include <hip/hip_bf16.h>
#include <math.h>

#define N_QUBITS 15
#define DIM      32768          // 2^15
#define N_PQC    60
#define N_TOK    32
#define D_MODEL  1024

typedef __attribute__((ext_vector_type(16))) _Float16 v16h;
typedef __attribute__((ext_vector_type(8)))  float    v8f;

// ---------------------------------------------------------------------------
// Small setup: normalized LCU coefficients + sum|qsvt|
// ---------------------------------------------------------------------------
__global__ void setup_kernel(const float* __restrict__ lcu_ri,
                             const float* __restrict__ qsvt,
                             float2* __restrict__ lcu,
                             float* __restrict__ scal) {
  if (threadIdx.x == 0 && blockIdx.x == 0) {
    float s = 0.f;
    for (int t = 0; t < N_TOK; ++t) {
      float re = lcu_ri[2 * t], im = lcu_ri[2 * t + 1];
      s += sqrtf(re * re + im * im);
    }
    float g = fmaxf(s, 1e-8f);
    for (int t = 0; t < N_TOK; ++t) {
      float2 c; c.x = lcu_ri[2 * t] / g; c.y = lcu_ri[2 * t + 1] / g;
      lcu[t] = c;
    }
    float qs = 0.f;
    for (int i = 0; i < 4; ++i) qs += fabsf(qsvt[i]);
    scal[0] = qs;
  }
}

// mono = e0 ; acc = qsvt[0] * e0
__global__ void init_kernel(const float* __restrict__ qsvt,
                            float2* __restrict__ mono,
                            float2* __restrict__ acc) {
  int i = blockIdx.x * blockDim.x + threadIdx.x;
  if (i >= DIM) return;
  float2 m; m.x = (i == 0) ? 1.f : 0.f; m.y = 0.f;
  float2 a; a.x = (i == 0) ? qsvt[0] : 0.f; a.y = 0.f;
  mono[i] = m;
  acc[i]  = a;
}

// ---------------------------------------------------------------------------
// Angle GEMM via WMMA: angles[32][64] = emb[32][1024] @ W^T[1024][60] + b
// One 256-thread block = 8 waves; wave handles one 16x16 output tile.
// All loads are unconditional 16B vector loads; OOB columns (n>=60) are
// clamped to row 0 and zero-selected in registers (no EXEC divergence).
// ---------------------------------------------------------------------------
__global__ __launch_bounds__(256)
void gemm_angles_kernel(const float* __restrict__ emb,
                        const float* __restrict__ W,
                        const float* __restrict__ bias,
                        float* __restrict__ angles) {
  const int lane = threadIdx.x & 31;
  const int wave = threadIdx.x >> 5;   // 0..7
  const int mt   = wave & 1;           // M tile (0..1)
  const int nt   = wave >> 1;          // N tile (0..3)
  const int h    = lane >> 4;          // lane half
  const int l16  = lane & 15;
  const int m    = mt * 16 + l16;      // token row, always < 32
  const int n    = nt * 16 + l16;      // param col, 0..63
  const bool nvalid = (n < N_PQC);
  const int nrow = nvalid ? n : 0;     // clamped: loads always in-bounds

  v8f c = {};
  for (int kk = 0; kk < D_MODEL / 32; ++kk) {
    const int koff = kk * 32;

    // ---- A operand: lane (m, half h) holds K = {h*8..h*8+7, 16+h*8..16+h*8+7}
    const float* arow = emb + m * D_MODEL + koff + h * 8;
    float4 a0 = ((const float4*)arow)[0];
    float4 a1 = ((const float4*)arow)[1];
    float4 a2 = ((const float4*)(arow + 16))[0];
    float4 a3 = ((const float4*)(arow + 16))[1];
    v16h a;
    a[0]  = (_Float16)a0.x; a[1]  = (_Float16)a0.y;
    a[2]  = (_Float16)a0.z; a[3]  = (_Float16)a0.w;
    a[4]  = (_Float16)a1.x; a[5]  = (_Float16)a1.y;
    a[6]  = (_Float16)a1.z; a[7]  = (_Float16)a1.w;
    a[8]  = (_Float16)a2.x; a[9]  = (_Float16)a2.y;
    a[10] = (_Float16)a2.z; a[11] = (_Float16)a2.w;
    a[12] = (_Float16)a3.x; a[13] = (_Float16)a3.y;
    a[14] = (_Float16)a3.z; a[15] = (_Float16)a3.w;

    // ---- B operand: lane holds column n, K = {h*8..h*8+7, 16+h*8..16+h*8+7}
    const float* wrow = W + nrow * D_MODEL + koff + h * 8;
    float4 w0 = ((const float4*)wrow)[0];
    float4 w1 = ((const float4*)wrow)[1];
    float4 w2 = ((const float4*)wrow)[4];
    float4 w3 = ((const float4*)wrow)[5];
    const float zf = 0.f;
    v16h bm;
    bm[0]  = (_Float16)(nvalid ? w0.x : zf); bm[1]  = (_Float16)(nvalid ? w0.y : zf);
    bm[2]  = (_Float16)(nvalid ? w0.z : zf); bm[3]  = (_Float16)(nvalid ? w0.w : zf);
    bm[4]  = (_Float16)(nvalid ? w1.x : zf); bm[5]  = (_Float16)(nvalid ? w1.y : zf);
    bm[6]  = (_Float16)(nvalid ? w1.z : zf); bm[7]  = (_Float16)(nvalid ? w1.w : zf);
    bm[8]  = (_Float16)(nvalid ? w2.x : zf); bm[9]  = (_Float16)(nvalid ? w2.y : zf);
    bm[10] = (_Float16)(nvalid ? w2.z : zf); bm[11] = (_Float16)(nvalid ? w2.w : zf);
    bm[12] = (_Float16)(nvalid ? w3.x : zf); bm[13] = (_Float16)(nvalid ? w3.y : zf);
    bm[14] = (_Float16)(nvalid ? w3.z : zf); bm[15] = (_Float16)(nvalid ? w3.w : zf);

    c = __builtin_amdgcn_wmma_f32_16x16x32_f16(false, a, false, bm,
                                               (short)0, c, false, false);
  }
#pragma unroll
  for (int v = 0; v < 8; ++v) {
    int row = mt * 16 + h * 8 + v;   // C/D layout: M = v + 8*half
    if (nvalid) angles[row * 64 + n] = c[v] + bias[n];
  }
}

// ---------------------------------------------------------------------------
// 60-gate PQC applied to LDS-resident state (ansatz 14, N_LAYERS=1)
// wire w <-> bit (14 - w)
// ---------------------------------------------------------------------------
__device__ __forceinline__ void apply_gates(float2* st, const float* __restrict__ ang) {
  for (int g = 0; g < N_PQC; ++g) {
    float theta = ang[g];
    float ch = cosf(0.5f * theta);
    float sh = sinf(0.5f * theta);
    int type, wc = 0, wt = 0;
    if (g < 15)      { type = 0; wc = g; }
    else if (g < 30) { type = 1; wc = 14 - (g - 15); wt = (wc + 1) % 15; }
    else if (g < 45) { type = 0; wc = g - 30; }
    else {
      int j = g - 45;
      type = 1; wc = (j == 0) ? 14 : (j - 1); wt = (wc + 14) % 15;
    }
    if (type == 0) {
      // RY on wire wc: [[c,-s],[s,c]]
      int b = 14 - wc;
      int stride = 1 << b;
      for (int p = threadIdx.x; p < (DIM >> 1); p += blockDim.x) {
        int i0 = ((p >> b) << (b + 1)) | (p & (stride - 1));
        int i1 = i0 | stride;
        float2 a0 = st[i0], a1 = st[i1];
        float2 n0, n1;
        n0.x = ch * a0.x - sh * a1.x;  n0.y = ch * a0.y - sh * a1.y;
        n1.x = sh * a0.x + ch * a1.x;  n1.y = sh * a0.y + ch * a1.y;
        st[i0] = n0; st[i1] = n1;
      }
    } else {
      // CRX ctrl=wc tgt=wt: where ctrl bit==1 apply [[c,-is],[-is,c]]
      int bc = 14 - wc, bt = 14 - wt;
      int lo = bc < bt ? bc : bt;
      int hi = bc < bt ? bt : bc;
      for (int p = threadIdx.x; p < (DIM >> 2); p += blockDim.x) {
        int t = ((p >> lo) << (lo + 1)) | (p & ((1 << lo) - 1));
        t = ((t >> hi) << (hi + 1)) | (t & ((1 << hi) - 1));
        int i0 = t | (1 << bc);          // ctrl=1, tgt=0
        int i1 = i0 | (1 << bt);         // ctrl=1, tgt=1
        float2 a0 = st[i0], a1 = st[i1];
        float2 n0, n1;
        n0.x = ch * a0.x + sh * a1.y;  n0.y = ch * a0.y - sh * a1.x;
        n1.x = ch * a1.x + sh * a0.y;  n1.y = ch * a1.y - sh * a0.x;
        st[i0] = n0; st[i1] = n1;
      }
    }
    __syncthreads();
  }
}

// ---------------------------------------------------------------------------
// norm of complex vector -> scalar slot
// ---------------------------------------------------------------------------
__global__ __launch_bounds__(1024)
void norm_kernel(const float2* __restrict__ v, float* __restrict__ outSlot) {
  __shared__ float red[1024];
  float s = 0.f;
  for (int i = threadIdx.x; i < DIM; i += 1024) {
    float2 a = v[i];
    s += a.x * a.x + a.y * a.y;
  }
  red[threadIdx.x] = s;
  __syncthreads();
  for (int o = 512; o > 0; o >>= 1) {
    if (threadIdx.x < o) red[threadIdx.x] += red[threadIdx.x + o];
    __syncthreads();
  }
  if (threadIdx.x == 0) *outSlot = sqrtf(red[0]);
}

// ---------------------------------------------------------------------------
// One block per token: normalize mono into LDS, evolve, write evolved[t]
// ---------------------------------------------------------------------------
__global__ __launch_bounds__(1024)
void evolve_tokens_kernel(const float2* __restrict__ mono,
                          const float* __restrict__ scal,
                          const float* __restrict__ angles,
                          float2* __restrict__ evolved) {
  extern __shared__ char smem[];
  float2* st = (float2*)smem;
  const int t = blockIdx.x;
  float nrm = scal[1];
  bool small = nrm < 1e-8f;
  float inv = small ? 0.f : (1.f / nrm);
  for (int i = threadIdx.x; i < DIM; i += blockDim.x) {
    float2 s;
    if (small) { s.x = (i == 0) ? 1.f : 0.f; s.y = 0.f; }
    else       { float2 v = mono[i]; s.x = v.x * inv; s.y = v.y * inv; }
    st[i] = s;
  }
  __syncthreads();
  apply_gates(st, angles + t * 64);
  float2* out = evolved + (size_t)t * DIM;
  for (int i = threadIdx.x; i < DIM; i += blockDim.x) out[i] = st[i];
}

// ---------------------------------------------------------------------------
// mono = restore * sum_t lcu[t]*evolved[t];  acc += qsvt[k]*mono
// ---------------------------------------------------------------------------
__global__ void lcu_reduce_kernel(const float2* __restrict__ evolved,
                                  const float2* __restrict__ lcu,
                                  const float* __restrict__ scal,
                                  const float* __restrict__ qsvt, int k,
                                  float2* __restrict__ mono,
                                  float2* __restrict__ acc) {
  int i = blockIdx.x * blockDim.x + threadIdx.x;
  if (i >= DIM) return;
  float2 sum; sum.x = 0.f; sum.y = 0.f;
  for (int t = 0; t < N_TOK; ++t) {
    if (t + 1 < N_TOK)
      __builtin_prefetch(&evolved[(size_t)(t + 1) * DIM + i], 0, 1);
    float2 e = evolved[(size_t)t * DIM + i];
    float2 c = lcu[t];
    sum.x += c.x * e.x - c.y * e.y;
    sum.y += c.x * e.y + c.y * e.x;
  }
  float nrm = scal[1];
  float restore = (nrm < 1e-8f) ? 1.f : nrm;
  float2 m; m.x = restore * sum.x; m.y = restore * sum.y;
  mono[i] = m;
  float q = qsvt[k];
  float2 a = acc[i];
  a.x += q * m.x; a.y += q * m.y;
  acc[i] = a;
}

// ---------------------------------------------------------------------------
// Final: normalize acc, evolve with ff_params, measure X/Y/Z per wire
// ---------------------------------------------------------------------------
__global__ __launch_bounds__(1024)
void final_kernel(const float2* __restrict__ acc,
                  const float* __restrict__ scal,
                  const float* __restrict__ ff,
                  float* __restrict__ out) {
  extern __shared__ char smem[];
  float2* st = (float2*)smem;
  float* rx = (float*)(smem + 2u * DIM * sizeof(float));
  float* ry = rx + 1024;
  float* rz = ry + 1024;

  float na = scal[2];                       // ||acc||
  float qg = fmaxf(scal[0], 1e-8f);         // sum|qsvt| guard
  float snorm = na / qg;                    // ||acc/qg||
  bool small = snorm < 1e-8f;
  float inv = small ? 0.f : (1.f / na);     // (acc/qg)/snorm == acc/||acc||
  for (int i = threadIdx.x; i < DIM; i += blockDim.x) {
    float2 s;
    if (small) { s.x = (i == 0) ? 1.f : 0.f; s.y = 0.f; }
    else       { float2 v = acc[i]; s.x = v.x * inv; s.y = v.y * inv; }
    st[i] = s;
  }
  __syncthreads();
  apply_gates(st, ff);

  for (int w = 0; w < N_QUBITS; ++w) {
    int b = 14 - w;
    int stride = 1 << b;
    float lx = 0.f, ly = 0.f, lz = 0.f;
    for (int p = threadIdx.x; p < (DIM >> 1); p += blockDim.x) {
      int i0 = ((p >> b) << (b + 1)) | (p & (stride - 1));
      int i1 = i0 | stride;
      float2 a0 = st[i0], a1 = st[i1];
      float re = a0.x * a1.x + a0.y * a1.y;   // Re(conj(a0)*a1)
      float im = a0.x * a1.y - a0.y * a1.x;   // Im(conj(a0)*a1)
      lx += 2.f * re;
      ly += 2.f * im;
      lz += (a0.x * a0.x + a0.y * a0.y) - (a1.x * a1.x + a1.y * a1.y);
    }
    rx[threadIdx.x] = lx; ry[threadIdx.x] = ly; rz[threadIdx.x] = lz;
    __syncthreads();
    for (int o = 512; o > 0; o >>= 1) {
      if (threadIdx.x < o) {
        rx[threadIdx.x] += rx[threadIdx.x + o];
        ry[threadIdx.x] += ry[threadIdx.x + o];
        rz[threadIdx.x] += rz[threadIdx.x + o];
      }
      __syncthreads();
    }
    if (threadIdx.x == 0) {
      out[w]      = rx[0];
      out[15 + w] = ry[0];
      out[30 + w] = rz[0];
    }
    __syncthreads();
  }
}

// ---------------------------------------------------------------------------
// Host launch
// ---------------------------------------------------------------------------
extern "C" void kernel_launch(void* const* d_in, const int* in_sizes, int n_in,
                              void* d_out, int out_size, void* d_ws, size_t ws_size,
                              hipStream_t stream) {
  const float* emb    = (const float*)d_in[0];   // [32,1024]
  const float* W      = (const float*)d_in[1];   // [60,1024]
  const float* bias   = (const float*)d_in[2];   // [60]
  const float* qsvt   = (const float*)d_in[3];   // [4]
  const float* lcu_ri = (const float*)d_in[4];   // [32,2]
  const float* ff     = (const float*)d_in[5];   // [60]
  float* out = (float*)d_out;                    // [45]

  float* ws = (float*)d_ws;
  float2* lcu    = (float2*)ws;                         // 32 complex
  float*  scal   = ws + 64;                             // [0]=qsum [1]=||mono|| [2]=||acc||
  float*  angles = ws + 128;                            // 32x64
  float2* mono   = (float2*)(ws + 2176);                // DIM complex
  float2* acc    = (float2*)(ws + 2176 + 2 * DIM);      // DIM complex
  float2* evolved= (float2*)(ws + 2176 + 4 * DIM);      // 32 x DIM complex

  setup_kernel<<<1, 32, 0, stream>>>(lcu_ri, qsvt, lcu, scal);
  init_kernel<<<DIM / 256, 256, 0, stream>>>(qsvt, mono, acc);
  gemm_angles_kernel<<<1, 256, 0, stream>>>(emb, W, bias, angles);

  const size_t ldsState = 2u * DIM * sizeof(float);     // 256 KB
  for (int k = 1; k < 4; ++k) {
    norm_kernel<<<1, 1024, 0, stream>>>(mono, scal + 1);
    evolve_tokens_kernel<<<N_TOK, 1024, ldsState, stream>>>(mono, scal, angles, evolved);
    lcu_reduce_kernel<<<DIM / 256, 256, 0, stream>>>(evolved, lcu, scal, qsvt, k, mono, acc);
  }

  norm_kernel<<<1, 1024, 0, stream>>>(acc, scal + 2);
  final_kernel<<<1, 1024, ldsState + 3u * 1024 * sizeof(float), stream>>>(acc, scal, ff, out);
}
